// PYG_GAT_50697793962363
// MI455X (gfx1250) — matrix operand order repaired and checked by reading, not compiled, not run on hardware
//
#include <hip/hip_runtime.h>

// ---------------------------------------------------------------------------
// Types for CDNA5 WMMA (wave32)
// ---------------------------------------------------------------------------
typedef __attribute__((ext_vector_type(16))) __bf16 v16bf;
typedef __attribute__((ext_vector_type(8)))  __bf16 v8bf;
typedef __attribute__((ext_vector_type(8)))  float  v8f;
typedef __attribute__((ext_vector_type(4)))  float  v4f;

#define EPS 1e-16f
#define SLOPE 0.2f

// ---------------------------------------------------------------------------
// fill kernel (zero-init for accumulators / atomic-max keys)
// ---------------------------------------------------------------------------
__global__ __launch_bounds__(256) void fill_u32(unsigned* __restrict__ p,
                                                unsigned v, int n) {
    int i = blockIdx.x * blockDim.x + threadIdx.x;
    if (i < n) p[i] = v;
}

// ---------------------------------------------------------------------------
// GEMM: H[M,Nout] = X[M,K] * W[Nout,K]^T  via V_WMMA_F32_16X16X32_BF16.
// K is a compile-time constant -> fully unrolled straight-line WMMA chain.
//
// Block = 256 threads (8 waves). Each block:
//   - stages a 64-row strip of W into LDS as bf16 (padded stride KP=K+8 so
//     the 16 column-lanes hit disjoint bank groups: conflict-free
//     ds_load_b128 pairs),
//   - each wave computes a 16x64 output strip (4 accumulator tiles, one
//     shared A fragment per K-chunk),
//   - A loads from global are software-pipelined; with full unroll the
//     compiler schedules them (and the B ds_loads) well ahead of use.
//
// Fragment layouts (wave32):
//   A (16x32 bf16): lane l holds row M=l&15; elems 0..7 -> K=kk+(l>>4)*8..+7,
//                   elems 8..15 -> K=kk+16+(l>>4)*8..+7
//   B (32x16 bf16): lane l holds col N=l&15; elems 0..7 -> K=kk+(l>>4)*8..+7, 8..15 -> K=kk+16+(l>>4)*8..+7
//   C/D (16x16 f32): VGPR r: M=r+(l<16?0:8), N=l&15
// ---------------------------------------------------------------------------
template <int KC>
__global__ __launch_bounds__(256) void gemm_wmma_bf16_k(
    const float* __restrict__ X, const float* __restrict__ W,
    float* __restrict__ H, int M, int Nout) {
    constexpr int KP = KC + 8;  // padded LDS row stride (bank spread)
    __shared__ __bf16 wlds[64 * KP];

    const int colBlocks = Nout >> 6;  // Nout / 64
    const int cb    = blockIdx.x % colBlocks;
    const int mb    = blockIdx.x / colBlocks;
    const int nBase = cb << 6;

    // ---- cooperative stage: W[nBase..nBase+63][0..KC) -> bf16 LDS ----
    for (int i = threadIdx.x; i < (KC << 6); i += 256) {
        int n = i / KC, k = i - n * KC;
        wlds[n * KP + k] = (__bf16)W[(size_t)(nBase + n) * KC + k];
    }
    __syncthreads();

    const int wave  = threadIdx.x >> 5;
    const int lane  = threadIdx.x & 31;
    const int mTile = mb * 8 + wave;
    if (mTile * 16 >= M) return;

    const int m0   = mTile * 16;
    const int half = lane >> 4;
    const int col  = lane & 15;

    v8f acc[4] = {};
    const float* arow = X + (size_t)(m0 + col) * KC;

    // prime the A pipeline
    v4f x0, x1, x2, x3;
    {
        const float* p0 = arow + half * 8;
        const float* p1 = arow + 16 + half * 8;
        x0 = *(const v4f*)p0;
        x1 = *(const v4f*)(p0 + 4);
        x2 = *(const v4f*)p1;
        x3 = *(const v4f*)(p1 + 4);
    }

#pragma unroll
    for (int kk = 0; kk < KC; kk += 32) {
        // convert staged A chunk -> bf16 fragment
        v16bf a;
#pragma unroll
        for (int t = 0; t < 4; ++t) {
            a[t]      = (__bf16)x0[t];
            a[4 + t]  = (__bf16)x1[t];
            a[8 + t]  = (__bf16)x2[t];
            a[12 + t] = (__bf16)x3[t];
        }
        // prefetch next A chunk (guard folds at compile time)
        if constexpr (true) {
            if (kk + 32 < KC) {
                const float* p0 = arow + kk + 32 + half * 8;
                const float* p1 = arow + kk + 48 + half * 8;
                x0 = *(const v4f*)p0;
                x1 = *(const v4f*)(p0 + 4);
                x2 = *(const v4f*)p1;
                x3 = *(const v4f*)(p1 + 4);
            }
        }
        // 4 output tiles share the A fragment; B comes from LDS
#pragma unroll
        for (int t = 0; t < 4; ++t) {
            const __bf16* bp = wlds + (t * 16 + col) * KP + kk + half * 8;
            v8bf b0 = *(const v8bf*)bp;
            v8bf b1 = *(const v8bf*)(bp + 16);
            v16bf b;
#pragma unroll
            for (int q = 0; q < 8; ++q) {
                b[q]     = b0[q];
                b[8 + q] = b1[q];
            }
            acc[t] = __builtin_amdgcn_wmma_f32_16x16x32_bf16(
                false, a, false, b, (short)0, acc[t], false, false);
        }
    }

    // ---- store C ----
#pragma unroll
    for (int t = 0; t < 4; ++t) {
#pragma unroll
        for (int r = 0; r < 8; ++r) {
            H[(size_t)(m0 + half * 8 + r) * Nout + nBase + t * 16 + col] =
                acc[t][r];
        }
    }
}

// runtime-K fallback (same structure, runtime loop)
__global__ __launch_bounds__(256) void gemm_wmma_bf16_rt(
    const float* __restrict__ X, const float* __restrict__ W,
    float* __restrict__ H, int M, int K, int Nout) {
    __shared__ __bf16 wlds[64 * 264];
    const int KP = K + 8;
    const int colBlocks = Nout >> 6;
    const int cb    = blockIdx.x % colBlocks;
    const int mb    = blockIdx.x / colBlocks;
    const int nBase = cb << 6;

    for (int i = threadIdx.x; i < (K << 6); i += 256) {
        int n = i / K, k = i - n * K;
        wlds[n * KP + k] = (__bf16)W[(size_t)(nBase + n) * K + k];
    }
    __syncthreads();

    const int wave  = threadIdx.x >> 5;
    const int lane  = threadIdx.x & 31;
    const int mTile = mb * 8 + wave;
    if (mTile * 16 >= M) return;

    const int m0   = mTile * 16;
    const int half = lane >> 4;
    const int col  = lane & 15;

    v8f acc[4] = {};
    const float* arow = X + (size_t)(m0 + col) * K;

    for (int kk = 0; kk < K; kk += 32) {
        const float* p0 = arow + kk + half * 8;
        const float* p1 = arow + kk + 16 + half * 8;
        v4f x0 = *(const v4f*)p0;
        v4f x1 = *(const v4f*)(p0 + 4);
        v4f x2 = *(const v4f*)p1;
        v4f x3 = *(const v4f*)(p1 + 4);
        v16bf a;
#pragma unroll
        for (int t = 0; t < 4; ++t) {
            a[t]      = (__bf16)x0[t];
            a[4 + t]  = (__bf16)x1[t];
            a[8 + t]  = (__bf16)x2[t];
            a[12 + t] = (__bf16)x3[t];
        }
#pragma unroll
        for (int t = 0; t < 4; ++t) {
            const __bf16* bp = wlds + (t * 16 + col) * KP + kk + half * 8;
            v8bf b0 = *(const v8bf*)bp;
            v8bf b1 = *(const v8bf*)(bp + 16);
            v16bf b;
#pragma unroll
            for (int q = 0; q < 8; ++q) {
                b[q]     = b0[q];
                b[8 + q] = b1[q];
            }
            acc[t] = __builtin_amdgcn_wmma_f32_16x16x32_bf16(
                false, a, false, b, (short)0, acc[t], false, false);
        }
    }
#pragma unroll
    for (int t = 0; t < 4; ++t) {
#pragma unroll
        for (int r = 0; r < 8; ++r) {
            H[(size_t)(m0 + half * 8 + r) * Nout + nBase + t * 16 + col] =
                acc[t][r];
        }
    }
}

// ---------------------------------------------------------------------------
// Per-node attention scalars: a_s[i] = h[i,:].att_src ; a_d[i] = h[i,:].att_dst
// One wave per row, shuffle reduction (wave32).
// ---------------------------------------------------------------------------
__global__ __launch_bounds__(256) void row_attn(
    const float* __restrict__ H, const float* __restrict__ atts,
    const float* __restrict__ attd, float* __restrict__ a_s,
    float* __restrict__ a_d, int N, int F) {
    const int wid  = (blockIdx.x * blockDim.x + threadIdx.x) >> 5;
    const int lane = threadIdx.x & 31;
    if (wid >= N) return;
    const float* hr = H + (size_t)wid * F;
    float s = 0.f, d = 0.f;
    for (int f = lane; f < F; f += 32) {
        float h = hr[f];
        s += h * atts[f];
        d += h * attd[f];
    }
#pragma unroll
    for (int m = 16; m > 0; m >>= 1) {
        s += __shfl_xor(s, m, 32);
        d += __shfl_xor(d, m, 32);
    }
    if (lane == 0) {
        a_s[wid] = s;
        a_d[wid] = d;
    }
}

// monotonic float<->uint key mapping (unsigned compare == float compare)
__device__ __forceinline__ unsigned f2key(float f) {
    unsigned b = __float_as_uint(f);
    return (b & 0x80000000u) ? ~b : (b | 0x80000000u);
}
__device__ __forceinline__ float key2f(unsigned k) {
    unsigned b = (k & 0x80000000u) ? (k & 0x7FFFFFFFu) : ~k;
    return __uint_as_float(b);
}

__device__ __forceinline__ void edge_sd(const long long* ei, int e, int E,
                                        int& s, int& d) {
    if (e < E) {
        s = (int)ei[e];
        d = (int)ei[(size_t)E + e];
    } else {
        s = d = e - E;  // self loop
    }
}

// ---------------------------------------------------------------------------
// Edge pass 1: segment max of leaky_relu(a_s[src]+a_d[dst]) per dst.
// ---------------------------------------------------------------------------
__global__ __launch_bounds__(256) void edge_max(
    const float* __restrict__ a_s, const float* __restrict__ a_d,
    const long long* __restrict__ ei, unsigned* __restrict__ emaxkey, int E,
    int N) {
    const int e = blockIdx.x * blockDim.x + threadIdx.x;
    if (e >= E + N) return;
    int s, d;
    edge_sd(ei, e, E, s, d);
    float v = a_s[s] + a_d[d];
    v = (v > 0.f) ? v : SLOPE * v;
    atomicMax(&emaxkey[d], f2key(v));
}

// ---------------------------------------------------------------------------
// Edge pass 2: ex = exp(e - emax[dst]); denom[dst] += ex; stash ex per edge.
// ---------------------------------------------------------------------------
__global__ __launch_bounds__(256) void edge_expsum(
    const float* __restrict__ a_s, const float* __restrict__ a_d,
    const long long* __restrict__ ei, const unsigned* __restrict__ emaxkey,
    float* __restrict__ exbuf, float* __restrict__ denom, int E, int N) {
    const int e = blockIdx.x * blockDim.x + threadIdx.x;
    if (e >= E + N) return;
    int s, d;
    edge_sd(ei, e, E, s, d);
    float v = a_s[s] + a_d[d];
    v = (v > 0.f) ? v : SLOPE * v;
    float ex = __expf(v - key2f(emaxkey[d]));
    exbuf[e] = ex;
    atomicAdd(&denom[d], ex);
}

// ---------------------------------------------------------------------------
// Edge pass 3: out[dst,:] += (ex / (denom[dst]+eps)) * h[src,:]
// One wave per edge; lane owns VEC contiguous features (vector load + per
// component global_atomic_add_f32, all L2 resident on MI455X).
// ---------------------------------------------------------------------------
template <int VEC>
__global__ __launch_bounds__(256) void edge_aggregate_vec(
    const float* __restrict__ H, const long long* __restrict__ ei,
    const float* __restrict__ exbuf, const float* __restrict__ denom,
    float* __restrict__ out, int E, int N, int F) {
    typedef float vf __attribute__((ext_vector_type(VEC)));
    const int wid  = (blockIdx.x * blockDim.x + threadIdx.x) >> 5;
    const int lane = threadIdx.x & 31;
    if (wid >= E + N) return;
    int s, d;
    edge_sd(ei, wid, E, s, d);
    const float alpha = exbuf[wid] / (denom[d] + EPS);
    const float* hr = H + (size_t)s * F + lane * VEC;
    float* orow     = out + (size_t)d * F + lane * VEC;
    vf hv = *(const vf*)hr;
#pragma unroll
    for (int i = 0; i < VEC; ++i) atomicAdd(&orow[i], alpha * hv[i]);
}

// generic fallback (F other than 64 or 128)
__global__ __launch_bounds__(256) void edge_aggregate_gen(
    const float* __restrict__ H, const long long* __restrict__ ei,
    const float* __restrict__ exbuf, const float* __restrict__ denom,
    float* __restrict__ out, int E, int N, int F) {
    const int wid  = (blockIdx.x * blockDim.x + threadIdx.x) >> 5;
    const int lane = threadIdx.x & 31;
    if (wid >= E + N) return;
    int s, d;
    edge_sd(ei, wid, E, s, d);
    const float alpha = exbuf[wid] / (denom[d] + EPS);
    const float* hr = H + (size_t)s * F;
    float* orow     = out + (size_t)d * F;
    for (int f = lane; f < F; f += 32) atomicAdd(&orow[f], alpha * hr[f]);
}

// ---------------------------------------------------------------------------
// out[i,j] += bias[j]
// ---------------------------------------------------------------------------
__global__ __launch_bounds__(256) void bias_add(float* __restrict__ out,
                                                const float* __restrict__ b,
                                                int total, int F) {
    int i = blockIdx.x * blockDim.x + threadIdx.x;
    if (i < total) out[i] += b[i % F];
}

// ---------------------------------------------------------------------------
// Host-side orchestration
// ---------------------------------------------------------------------------
static inline int cdiv(long long a, long long b) { return (int)((a + b - 1) / b); }

static void run_layer(const float* Xin, int N, int K, int F, int E,
                      const long long* ei, const float* Wm, const float* atts,
                      const float* attd, const float* bias, float* hfeat,
                      float* out, float* a_s, float* a_d, unsigned* emaxkey,
                      float* denom, float* exbuf, hipStream_t stream) {
    const int ET = E + N;
    // GEMM: hfeat = Xin * Wm^T  (block: 8 M-tiles x 64 cols, W staged in LDS)
    {
        const int colBlocks = F / 64;
        const int mBlocks   = cdiv(N / 16, 8);
        const int blocks    = mBlocks * colBlocks;
        if (K == 256)
            gemm_wmma_bf16_k<256><<<blocks, 256, 0, stream>>>(Xin, Wm, hfeat, N,
                                                              F);
        else if (K == 128)
            gemm_wmma_bf16_k<128><<<blocks, 256, 0, stream>>>(Xin, Wm, hfeat, N,
                                                              F);
        else
            gemm_wmma_bf16_rt<<<blocks, 256, 0, stream>>>(Xin, Wm, hfeat, N, K,
                                                          F);
    }
    // attention scalars
    row_attn<<<cdiv(N, 8), 256, 0, stream>>>(hfeat, atts, attd, a_s, a_d, N, F);
    // clear softmax state + output accumulator
    fill_u32<<<cdiv(N, 256), 256, 0, stream>>>(emaxkey, 0u, N);
    fill_u32<<<cdiv(N, 256), 256, 0, stream>>>((unsigned*)denom, 0u, N);
    fill_u32<<<cdiv((long long)N * F, 256), 256, 0, stream>>>((unsigned*)out, 0u,
                                                              N * F);
    // edge softmax + aggregation
    edge_max<<<cdiv(ET, 256), 256, 0, stream>>>(a_s, a_d, ei, emaxkey, E, N);
    edge_expsum<<<cdiv(ET, 256), 256, 0, stream>>>(a_s, a_d, ei, emaxkey, exbuf,
                                                   denom, E, N);
    {
        const int blocks = cdiv((long long)ET * 32, 256);
        if (F == 128)
            edge_aggregate_vec<4><<<blocks, 256, 0, stream>>>(
                hfeat, ei, exbuf, denom, out, E, N, F);
        else if (F == 64)
            edge_aggregate_vec<2><<<blocks, 256, 0, stream>>>(
                hfeat, ei, exbuf, denom, out, E, N, F);
        else
            edge_aggregate_gen<<<blocks, 256, 0, stream>>>(hfeat, ei, exbuf,
                                                           denom, out, E, N, F);
    }
    bias_add<<<cdiv((long long)N * F, 256), 256, 0, stream>>>(out, bias, N * F,
                                                              F);
}

extern "C" void kernel_launch(void* const* d_in, const int* in_sizes, int n_in,
                              void* d_out, int out_size, void* d_ws,
                              size_t ws_size, hipStream_t stream) {
    // inputs: x, edge_index, W1, as1, ad1, b1, W2, as2, ad2, b2
    const float*     x   = (const float*)d_in[0];
    const long long* ei  = (const long long*)d_in[1];
    const float*     W1  = (const float*)d_in[2];
    const float*     as1 = (const float*)d_in[3];
    const float*     ad1 = (const float*)d_in[4];
    const float*     b1  = (const float*)d_in[5];
    const float*     W2  = (const float*)d_in[6];
    const float*     as2 = (const float*)d_in[7];
    const float*     ad2 = (const float*)d_in[8];
    const float*     b2  = (const float*)d_in[9];

    const int Fh  = in_sizes[3];        // n_hidden = 128
    const int Fo  = in_sizes[7];        // out_feats = 64
    const int Fin = in_sizes[2] / Fh;   // in_feats = 256
    const int N   = in_sizes[0] / Fin;  // 100000
    const int E   = in_sizes[1] / 2;    // 1600000

    // workspace carve-up (256B aligned)
    char*  base = (char*)d_ws;
    size_t off  = 0;
    auto carve = [&](size_t bytes) {
        void* p = base + off;
        off += (bytes + 255) & ~(size_t)255;
        return p;
    };
    float*    hfeat   = (float*)carve((size_t)N * Fh * 4);  // reused for layer2
    float*    out1    = (float*)carve((size_t)N * Fh * 4);
    float*    a_s     = (float*)carve((size_t)N * 4);
    float*    a_d     = (float*)carve((size_t)N * 4);
    unsigned* emaxkey = (unsigned*)carve((size_t)N * 4);
    float*    denom   = (float*)carve((size_t)N * 4);
    float*    exbuf   = (float*)carve((size_t)(E + N) * 4);
    (void)ws_size;
    (void)n_in;
    (void)out_size;

    // layer 1: x[N,256] -> out1[N,128]
    run_layer(x, N, Fin, Fh, E, ei, W1, as1, ad1, b1, hfeat, out1, a_s, a_d,
              emaxkey, denom, exbuf, stream);
    // layer 2: out1[N,128] -> d_out[N,64]   (hfeat buffer reused)
    run_layer(out1, N, Fh, Fo, E, ei, W2, as2, ad2, b2, hfeat, (float*)d_out,
              a_s, a_d, emaxkey, denom, exbuf, stream);
}
